// SelfAttentionHead_55422257988351
// MI455X (gfx1250) — matrix operand order repaired and checked by reading, not compiled, not run on hardware
//
#include <hip/hip_runtime.h>
#include <hip/hip_bf16.h>
#include <stdint.h>

#define S   4096
#define E   1024
#define NE3 3072

typedef __attribute__((ext_vector_type(16))) __bf16 v16bf;
typedef __attribute__((ext_vector_type(8)))  __bf16 v8bf;
typedef __attribute__((ext_vector_type(8)))  float  v8f;

static __device__ __forceinline__ __bf16 f2bf(float f) {
  union { float f; uint32_t u; } v; v.f = f;
  uint32_t r = v.u + 0x7FFFu + ((v.u >> 16) & 1u);   // round-to-nearest-even
  unsigned short h = (unsigned short)(r >> 16);
  return __builtin_bit_cast(__bf16, h);
}

// ---- WMMA fragment loaders (CDNA5 ISA 7.12.2, wave32) -----------------------
// A: 16x32 bf16 tile at (row0,k0) of a row-major matrix, leading dim lda.
// Lane L: M=L&15; K = 16*(L>>4)+i for halves i=0..15.
static __device__ __forceinline__ v16bf load_a(const __bf16* base, int lda,
                                               int row0, int k0, int lane) {
  const int m  = lane & 15;
  const int kb = (lane >> 4) * 16;
  const __bf16* p = base + (size_t)(row0 + m) * lda + k0 + kb;
  union { v16bf v; v8bf h[2]; } u;
  u.h[0] = *(const v8bf*)(p);
  u.h[1] = *(const v8bf*)(p + 8);
  return u.v;
}

// B: 32x16 tile where B[kk][n] = src[(col0+n)*lds + k0+kk] (column-contiguous
// source). Lane L: N=L&15; K = 16*(L>>4)+i for halves i=0..15. One 32B run.
static __device__ __forceinline__ v16bf load_b(const __bf16* base, int lds_,
                                               int col0, int k0, int lane) {
  const int n  = lane & 15;
  const int kb = (lane >> 4) * 16;
  const __bf16* p = base + (size_t)(col0 + n) * lds_ + k0 + kb;
  union { v16bf v; v8bf h[2]; } u;
  u.h[0] = *(const v8bf*)(p);
  u.h[1] = *(const v8bf*)(p + 8);
  return u.v;
}

static __device__ __forceinline__ v8f wmma_bf16(v16bf a, v16bf b, v8f c) {
  return __builtin_amdgcn_wmma_f32_16x16x32_bf16(false, a, false, b,
                                                 (short)0, c, false, false);
}

// ---- conversion kernels -----------------------------------------------------
__global__ void __launch_bounds__(256) k_cvt_x(const float* __restrict__ x,
                                               __bf16* __restrict__ xb) {
  size_t i = (size_t)blockIdx.x * 256 + threadIdx.x;
  if (i < (size_t)S * E) xb[i] = f2bf(x[i]);
}

// Wt[n][k] = W[k][n]  (GEMM B-operand columns become contiguous)
__global__ void __launch_bounds__(256) k_cvt_wt(const float* __restrict__ W,
                                                __bf16* __restrict__ Wt) {
  size_t i = (size_t)blockIdx.x * 256 + threadIdx.x;
  if (i >= (size_t)NE3 * E) return;
  int n = (int)(i / E), k = (int)(i % E);
  Wt[i] = f2bf(W[(size_t)k * NE3 + n]);
}

// ---- QKV projection: qkv = x @ W + b ---------------------------------------
// 4 waves/block as 2x2; each wave computes a 32x64 macro-tile (2x4 WMMA tiles).
// grid (S/64, NE3/128)
__global__ void __launch_bounds__(128)
k_qkv(const __bf16* __restrict__ xb, const __bf16* __restrict__ Wt,
      const float* __restrict__ bias,
      __bf16* __restrict__ qb, __bf16* __restrict__ kbuf, __bf16* __restrict__ vT) {
  const int lane = threadIdx.x & 31;
  const int wave = __builtin_amdgcn_readfirstlane(threadIdx.x >> 5); // SGPR
  const int row0 = blockIdx.x * 64 + (wave >> 1) * 32;               // uniform
  const int col0 = blockIdx.y * 128 + (wave & 1) * 64;               // uniform
  v8f acc[2][4] = {};
  for (int k0 = 0; k0 < E; k0 += 32) {
    v16bf a[2], b[4];
#pragma unroll
    for (int i = 0; i < 2; ++i) a[i] = load_a(xb, E, row0 + 16 * i, k0, lane);
#pragma unroll
    for (int j = 0; j < 4; ++j) b[j] = load_b(Wt, E, col0 + 16 * j, k0, lane);
#pragma unroll
    for (int i = 0; i < 2; ++i)
#pragma unroll
      for (int j = 0; j < 4; ++j) acc[i][j] = wmma_bf16(a[i], b[j], acc[i][j]);
  }
  const int nlo = lane & 15;
  const int rhi = (lane >> 4) << 3;
#pragma unroll
  for (int j = 0; j < 4; ++j) {
    const int cb   = col0 + 16 * j;          // wave-uniform segment selector
    const int colg = cb + nlo;
    const float bv = bias[colg];
    if (cb < E) {                            // scalar branch: Q segment
#pragma unroll
      for (int i = 0; i < 2; ++i)
#pragma unroll
        for (int r = 0; r < 8; ++r)
          qb[(size_t)(row0 + 16 * i + rhi + r) * E + colg] =
              f2bf(acc[i][j][r] + bv);
    } else if (cb < 2 * E) {                 // scalar branch: K segment
#pragma unroll
      for (int i = 0; i < 2; ++i)
#pragma unroll
        for (int r = 0; r < 8; ++r)
          kbuf[(size_t)(row0 + 16 * i + rhi + r) * E + (colg - E)] =
              f2bf(acc[i][j][r] + bv);
    } else {                                 // scalar branch: V (transposed)
#pragma unroll
      for (int i = 0; i < 2; ++i)
#pragma unroll
        for (int r = 0; r < 8; ++r)
          vT[(size_t)(colg - 2 * E) * S + (row0 + 16 * i + rhi + r)] =
              f2bf(acc[i][j][r] + bv);
    }
  }
}

// ---- scores = (q @ k^T) / sqrt(E), causal ----------------------------------
// grid (S/64, S/128); scalar (SGPR) skip of tiles strictly above the diagonal.
__global__ void __launch_bounds__(128)
k_scores(const __bf16* __restrict__ qb, const __bf16* __restrict__ kbuf,
         float* __restrict__ sc) {
  const int lane = threadIdx.x & 31;
  const int wave = __builtin_amdgcn_readfirstlane(threadIdx.x >> 5);
  const int row0 = blockIdx.x * 64 + (wave >> 1) * 32;
  const int col0 = blockIdx.y * 128 + (wave & 1) * 64;
  if (col0 > row0 + 31) return;              // scalar branch: EXEC stays all-1s
  v8f acc[2][4] = {};
  for (int k0 = 0; k0 < E; k0 += 32) {
    v16bf a[2], b[4];
#pragma unroll
    for (int i = 0; i < 2; ++i) a[i] = load_a(qb, E, row0 + 16 * i, k0, lane);
#pragma unroll
    for (int j = 0; j < 4; ++j) b[j] = load_b(kbuf, E, col0 + 16 * j, k0, lane);
#pragma unroll
    for (int i = 0; i < 2; ++i)
#pragma unroll
      for (int j = 0; j < 4; ++j) acc[i][j] = wmma_bf16(a[i], b[j], acc[i][j]);
  }
  const int nlo = lane & 15;
  const int rhi = (lane >> 4) << 3;
#pragma unroll
  for (int i = 0; i < 2; ++i)
#pragma unroll
    for (int j = 0; j < 4; ++j) {
      const int colg = col0 + 16 * j + nlo;
#pragma unroll
      for (int r = 0; r < 8; ++r) {
        const int row = row0 + 16 * i + rhi + r;
        float v = acc[i][j][r] * 0.03125f;   // 1/sqrt(1024)
        if (colg > row) v = -3.0e38f;        // causal mask (v_cndmask)
        sc[(size_t)row * S + colg] = v;
      }
    }
}

// ---- row softmax -> bf16 P, zero padding past the causal edge --------------
__global__ void __launch_bounds__(256)
k_softmax(const float* __restrict__ sc, __bf16* __restrict__ P) {
  const int row = blockIdx.x;
  const int tid = threadIdx.x;
  const int len = row + 1;
  __shared__ float red[256];
  const float* src = sc + (size_t)row * S;
  float m = -3.0e38f;
  for (int c = tid; c < len; c += 256) m = fmaxf(m, src[c]);
  red[tid] = m; __syncthreads();
  for (int s = 128; s > 0; s >>= 1) {
    if (tid < s) red[tid] = fmaxf(red[tid], red[tid + s]);
    __syncthreads();
  }
  const float mx = red[0]; __syncthreads();
  float sum = 0.f;
  for (int c = tid; c < len; c += 256) sum += __expf(src[c] - mx);
  red[tid] = sum; __syncthreads();
  for (int s = 128; s > 0; s >>= 1) {
    if (tid < s) red[tid] += red[tid + s];
    __syncthreads();
  }
  const float inv = 1.0f / red[0];
  __bf16* dst = P + (size_t)row * S;
  for (int c = tid; c < len; c += 256) dst[c] = f2bf(__expf(src[c] - mx) * inv);
  // zero past the causal edge so P@V K-tiles straddling the diagonal read zeros
  const int pad_end0 = ((row / 32) + 3) * 32;
  const int pad_end  = pad_end0 < S ? pad_end0 : S;
  for (int c = len + tid; c < pad_end; c += 256) dst[c] = f2bf(0.0f);
}

// ---- out = P @ V  (K loop clipped at the causal boundary) ------------------
// grid (S/64, E/128)
__global__ void __launch_bounds__(128)
k_out(const __bf16* __restrict__ P, const __bf16* __restrict__ vT,
      float* __restrict__ out) {
  const int lane = threadIdx.x & 31;
  const int wave = __builtin_amdgcn_readfirstlane(threadIdx.x >> 5);
  const int row0 = blockIdx.x * 64 + (wave >> 1) * 32;
  const int e0   = blockIdx.y * 128 + (wave & 1) * 64;
  const int nk   = row0 / 32 + 1;            // K-chunks up to causal edge (SGPR)
  v8f acc[2][4] = {};
  for (int kb = 0; kb < nk; ++kb) {
    const int k0 = kb * 32;
    v16bf a[2], b[4];
#pragma unroll
    for (int i = 0; i < 2; ++i) a[i] = load_a(P, S, row0 + 16 * i, k0, lane);
#pragma unroll
    for (int j = 0; j < 4; ++j) b[j] = load_b(vT, S, e0 + 16 * j, k0, lane);
#pragma unroll
    for (int i = 0; i < 2; ++i)
#pragma unroll
      for (int j = 0; j < 4; ++j) acc[i][j] = wmma_bf16(a[i], b[j], acc[i][j]);
  }
  const int nlo = lane & 15;
  const int rhi = (lane >> 4) << 3;
#pragma unroll
  for (int i = 0; i < 2; ++i)
#pragma unroll
    for (int j = 0; j < 4; ++j) {
      const int colg = e0 + 16 * j + nlo;
#pragma unroll
      for (int r = 0; r < 8; ++r)
        out[(size_t)(row0 + 16 * i + rhi + r) * E + colg] = acc[i][j][r];
    }
}

extern "C" void kernel_launch(void* const* d_in, const int* in_sizes, int n_in,
                              void* d_out, int out_size, void* d_ws, size_t ws_size,
                              hipStream_t stream) {
  (void)in_sizes; (void)n_in; (void)out_size; (void)ws_size;
  const float* x    = (const float*)d_in[0];
  const float* W    = (const float*)d_in[1];
  const float* bias = (const float*)d_in[2];
  float* out = (float*)d_out;

  char* ws = (char*)d_ws;
  size_t off = 0;
  __bf16* xb   = (__bf16*)(ws + off); off += (size_t)S   * E * 2;  //  8 MB
  __bf16* Wt   = (__bf16*)(ws + off); off += (size_t)NE3 * E * 2;  //  6 MB
  __bf16* qb   = (__bf16*)(ws + off); off += (size_t)S   * E * 2;  //  8 MB
  __bf16* kbuf = (__bf16*)(ws + off); off += (size_t)S   * E * 2;  //  8 MB
  __bf16* vT   = (__bf16*)(ws + off); off += (size_t)E   * S * 2;  //  8 MB
  float*  sc   = (float*) (ws + off); off += (size_t)S   * S * 4;  // 64 MB
  __bf16* P    = (__bf16*)(ws + off); off += (size_t)S   * S * 2;  // 32 MB

  k_cvt_x  <<<(S * E) / 256,   256, 0, stream>>>(x, xb);
  k_cvt_wt <<<(NE3 * E) / 256, 256, 0, stream>>>(W, Wt);
  k_qkv    <<<dim3(S / 64, NE3 / 128), 128, 0, stream>>>(xb, Wt, bias, qb, kbuf, vT);
  k_scores <<<dim3(S / 64, S / 128),   128, 0, stream>>>(qb, kbuf, sc);
  k_softmax<<<S,                       256, 0, stream>>>(sc, P);
  k_out    <<<dim3(S / 64, E / 128),   128, 0, stream>>>(P, vT, out);
}
